// Transformer_63866163691778
// MI455X (gfx1250) — matrix operand order repaired and checked by reading, not compiled, not run on hardware
//
#include <hip/hip_runtime.h>
#include <hip/hip_bf16.h>

// ---------------------------------------------------------------------------
// Transformer encoder-decoder forward (B=8, S=512, D=512, H=8, L=6, DFF=2048)
// Heavy GEMMs (~99.9% of FLOPs): v_wmma_f32_16x16x32_bf16, full tiles only,
// branch-free dword loads. Decoder fringes (seq len 1): scalar-dot kernel
// (the right tool for M=8 -- WMMA would waste >=50% of each tile).
// ---------------------------------------------------------------------------

#define D_MODEL 512
#define NHEAD   8
#define NLAYER  6
#define DFF_DIM 2048
#define OUT_DIM 1000
#define BATCH   8
#define SEQ     512
#define DK      64

typedef __attribute__((ext_vector_type(16))) __bf16 v16bf;
typedef __attribute__((ext_vector_type(8)))  float  v8f;

__device__ __forceinline__ unsigned short f2bf(float f) {
  unsigned int u = __float_as_uint(f);
  u += 0x7FFFu + ((u >> 16) & 1u);          // round-to-nearest-even
  return (unsigned short)(u >> 16);
}
__device__ __forceinline__ float bf2f(unsigned short h) {
  return __uint_as_float((unsigned int)h << 16);
}

// ---------------------------------------------------------------------------
// Full-tile batched WMMA GEMM: C[z][m][n] = alpha*sum_k A[z][m][k]*Bt[z][n][k]
// REQUIRES: M%16==0 (grid covers M), N%64==0, K%32==0, B in [N,K] layout.
//   A  bf16: A[aBatch*z + m*lda + k]
//   Bt bf16: B[bBatch*z + n*ldb + k]
//   Store: cT=0 -> Cf/Cbf[cBatch*z + m*ldc + n]
//          cT=1 -> Cbf[cBatch*z + n*ldc + m]   (transposed bf16, e.g. V^T)
// Wave tile 16x64 (4 x 16x16x32 WMMA per K-step), 4 waves/block -> 64x64.
// No bounds checks anywhere: pure vmem + wmma hot loop.
// ---------------------------------------------------------------------------
__global__ __launch_bounds__(128) void wmma_gemm_full_kernel(
    const unsigned short* __restrict__ A, long lda, long aBatch,
    const unsigned short* __restrict__ B, long ldb, long bBatch,
    float* __restrict__ Cf, unsigned short* __restrict__ Cbf, long ldc, long cBatch,
    int K, float alpha, const float* __restrict__ bias, int relu, int cT)
{
  const int batch = blockIdx.z;
  const int lane  = threadIdx.x & 31;
  const int wave  = threadIdx.x >> 5;
  const int m0    = blockIdx.y * 64 + wave * 16;
  const int n0    = blockIdx.x * 64;
  const int lhalf = lane >> 4;               // 0: lanes 0-15, 1: lanes 16-31
  const int lmod  = lane & 15;

  v8f acc[4];
  #pragma unroll
  for (int t = 0; t < 4; ++t)
    #pragma unroll
    for (int r = 0; r < 8; ++r) acc[t][r] = 0.0f;

  // lane-half K offsets baked into base pointers (A layout: +16, B layout: +16)
  const unsigned short* Arow = A + (long)batch * aBatch
                                 + (long)(m0 + lmod) * lda + (lhalf ? 16 : 0);
  const unsigned short* Br0  = B + (long)batch * bBatch
                                 + (long)(n0 + lmod) * ldb + (lhalf ? 16 : 0);

  for (int k0 = 0; k0 < K; k0 += 32) {
    __builtin_prefetch(Arow + k0 + 64, 0, 1);          // global_prefetch_b8
    union { unsigned int u[8]; v16bf v; } af;
    #pragma unroll
    for (int j = 0; j < 8; ++j) {
      const int kk = k0 + 2 * j;
      af.u[j] = *(const unsigned int*)(Arow + kk);
    }
    #pragma unroll
    for (int t = 0; t < 4; ++t) {
      const unsigned short* Brow = Br0 + (long)(t * 16) * ldb;
      union { unsigned int u[8]; v16bf v; } bf_;
      #pragma unroll
      for (int j = 0; j < 8; ++j)
        bf_.u[j] = *(const unsigned int*)(Brow + k0 + 2 * j);
      acc[t] = __builtin_amdgcn_wmma_f32_16x16x32_bf16(
          false, af.v, false, bf_.v, (short)0, acc[t], false, false);
    }
  }

  // epilogue: scale, bias, relu, store (unguarded -- tiles always full)
  #pragma unroll
  for (int t = 0; t < 4; ++t) {
    #pragma unroll
    for (int r = 0; r < 8; ++r) {
      const int m = m0 + r + 8 * lhalf;        // C/D layout: vgpr r -> M=r / r+8
      const int n = n0 + t * 16 + lmod;        // N striped across lanes
      float c = acc[t][r] * alpha;
      if (bias) c += bias[n];
      if (relu) c = fmaxf(c, 0.0f);
      if (cT) {
        Cbf[(long)batch * cBatch + (long)n * ldc + m] = f2bf(c);
      } else {
        const long idx = (long)batch * cBatch + (long)m * ldc + n;
        if (Cf)  Cf[idx]  = c;
        if (Cbf) Cbf[idx] = f2bf(c);
      }
    }
  }
}

// ---------------------------------------------------------------------------
// Small GEMM (decoder fringes, M<=8 / ragged N): one thread per output,
// fp32 FMA dot over bf16 operands. Optimal for skinny matrices.
// ---------------------------------------------------------------------------
__global__ __launch_bounds__(128) void small_gemm_kernel(
    const unsigned short* __restrict__ A, long lda, long aBatch,
    const unsigned short* __restrict__ B, long ldb, long bBatch, int bTrans,
    float* __restrict__ Cf, unsigned short* __restrict__ Cbf, long ldc, long cBatch,
    int M, int N, int K, float alpha, const float* __restrict__ bias, int relu)
{
  const int batch = blockIdx.z;
  const long idx = (long)blockIdx.x * blockDim.x + threadIdx.x;
  if (idx >= (long)M * N) return;
  const int m = (int)(idx / N);
  const int n = (int)(idx % N);
  const unsigned short* a = A + (long)batch * aBatch + (long)m * lda;
  const unsigned short* b = B + (long)batch * bBatch;
  float acc = 0.0f;
  if (bTrans) {
    const unsigned short* br = b + (long)n * ldb;
    for (int k = 0; k < K; ++k) acc += bf2f(a[k]) * bf2f(br[k]);
  } else {
    for (int k = 0; k < K; ++k) acc += bf2f(a[k]) * bf2f(b[(long)k * ldb + n]);
  }
  float c = acc * alpha;
  if (bias) c += bias[n];
  if (relu) c = fmaxf(c, 0.0f);
  const long o = (long)batch * cBatch + (long)m * ldc + n;
  if (Cf)  Cf[o]  = c;
  if (Cbf) Cbf[o] = f2bf(c);
}

// fp32 [K,N] row-major -> bf16 transposed [N,K]  (weight pre-pack)
__global__ __launch_bounds__(256) void convert_wt_kernel(
    const float* __restrict__ W, unsigned short* __restrict__ Wt, int Kd, int Nd)
{
  long idx = (long)blockIdx.x * blockDim.x + threadIdx.x;
  if (idx < (long)Kd * Nd) {
    int n = (int)(idx / Kd), k = (int)(idx % Kd);
    Wt[idx] = f2bf(W[(long)k * Nd + n]);
  }
}

// wave-per-row masked softmax (reference quirk: mask where score == 0.0)
__global__ __launch_bounds__(128) void softmax_mask_kernel(
    const float* __restrict__ S, unsigned short* __restrict__ P,
    int rows, int cols, int sq, int causal)
{
  int row = blockIdx.x * 4 + (threadIdx.x >> 5);
  if (row >= rows) return;
  int lane = threadIdx.x & 31;
  const float* s = S + (long)row * cols;
  unsigned short* p = P + (long)row * cols;
  int q = row % sq;
  float vals[16];
  int nchunk = (cols + 31) / 32;
  float mx = -3.4e38f;
  for (int i = 0; i < nchunk; ++i) {
    int c = i * 32 + lane;
    float v = -3.4e38f;
    if (c < cols) {
      v = s[c];
      bool keep = (v != 0.0f) && (!causal || c <= q);
      v = keep ? v : -1.0e9f;
    }
    vals[i] = v;
    mx = fmaxf(mx, v);
  }
  for (int off = 16; off >= 1; off >>= 1) mx = fmaxf(mx, __shfl_xor(mx, off, 32));
  float sum = 0.0f;
  for (int i = 0; i < nchunk; ++i) {
    int c = i * 32 + lane;
    float e = (c < cols) ? __expf(vals[i] - mx) : 0.0f;
    vals[i] = e; sum += e;
  }
  for (int off = 16; off >= 1; off >>= 1) sum += __shfl_xor(sum, off, 32);
  float inv = 1.0f / sum;
  for (int i = 0; i < nchunk; ++i) {
    int c = i * 32 + lane;
    if (c < cols) p[c] = f2bf(vals[i] * inv);
  }
}

// out = LayerNorm(a + b) * g + beta ; wave per row of 512 (no restrict: may alias out)
__global__ __launch_bounds__(128) void ln_residual_kernel(
    const float* A, const float* Bv, const float* G, const float* Be,
    float* Of, unsigned short* Obf, int rows)
{
  const int cols = D_MODEL;
  int row = blockIdx.x * 4 + (threadIdx.x >> 5);
  if (row >= rows) return;
  int lane = threadIdx.x & 31;
  long base = (long)row * cols;
  float v[16]; float sum = 0.0f;
  #pragma unroll
  for (int i = 0; i < 16; ++i) { int c = i * 32 + lane; float x = A[base + c] + Bv[base + c]; v[i] = x; sum += x; }
  for (int off = 16; off >= 1; off >>= 1) sum += __shfl_xor(sum, off, 32);
  float mean = sum * (1.0f / cols);
  float var = 0.0f;
  #pragma unroll
  for (int i = 0; i < 16; ++i) { float d = v[i] - mean; var += d * d; }
  for (int off = 16; off >= 1; off >>= 1) var += __shfl_xor(var, off, 32);
  var *= (1.0f / cols);
  float rstd = rsqrtf(var + 1e-5f);
  #pragma unroll
  for (int i = 0; i < 16; ++i) {
    int c = i * 32 + lane;
    float o = (v[i] - mean) * rstd * G[c] + Be[c];
    Of[base + c] = o; Obf[base + c] = f2bf(o);
  }
}

// encoder embed: e = in_emb[x]*sqrt(D) + PE[s]  (PE synthesized in-kernel)
__global__ __launch_bounds__(128) void enc_embed_kernel(
    const int* __restrict__ x, const float* __restrict__ emb,
    float* __restrict__ e, unsigned short* __restrict__ ebf)
{
  int row = blockIdx.x;                 // b*SEQ + s
  int s = row % SEQ;
  int tok = x[row];
  for (int c = threadIdx.x; c < D_MODEL; c += blockDim.x) {
    int cEven = c & ~1;
    float div = __expf((float)cEven * -0.017988946039016f);  // -ln(10000)/512
    float ang = (float)s * div;
    float pe  = (c & 1) ? __cosf(ang) : __sinf(ang);
    float v = emb[(long)tok * D_MODEL + c] * 22.62741699796952f + pe;
    e[(long)row * D_MODEL + c] = v;
    ebf[(long)row * D_MODEL + c] = f2bf(v);
  }
}

// decoder embed: d = out_emb[target]*sqrt(D) + PE[0]  (PE[0]: sin0=0, cos0=1)
__global__ __launch_bounds__(128) void dec_embed_kernel(
    const int* __restrict__ tgt, const float* __restrict__ emb,
    float* __restrict__ d, unsigned short* __restrict__ dbf)
{
  int b = blockIdx.x;
  int tok = tgt[b];
  for (int c = threadIdx.x; c < D_MODEL; c += blockDim.x) {
    float pe = (c & 1) ? 1.0f : 0.0f;
    float v = emb[(long)tok * D_MODEL + c] * 22.62741699796952f + pe;
    d[(long)b * D_MODEL + c] = v;
    dbf[(long)b * D_MODEL + c] = f2bf(v);
  }
}

// ---------------------------------------------------------------------------
// host-side helpers
// ---------------------------------------------------------------------------
static void launch_full(hipStream_t st,
                        const unsigned short* A, long lda, long aB,
                        const unsigned short* B, long ldb, long bB,
                        float* Cf, unsigned short* Cbf, long ldc, long cB,
                        int M, int N, int K, float alpha, const float* bias,
                        int relu, int batches, int cT = 0)
{
  // requires M%16==0 (M%64 not handled: grid.y = M/64 drops leftover rows,
  // no m0>=M guard -- callers here always have M%64==0), N%64==0, K%32==0
  dim3 grid(N / 64, M / 64, batches);
  wmma_gemm_full_kernel<<<grid, 128, 0, st>>>(A, lda, aB, B, ldb, bB,
                                              Cf, Cbf, ldc, cB, K, alpha, bias, relu, cT);
}

static void launch_small(hipStream_t st,
                         const unsigned short* A, long lda, long aB,
                         const unsigned short* B, long ldb, long bB, int bt,
                         float* Cf, unsigned short* Cbf, long ldc, long cB,
                         int M, int N, int K, float alpha, const float* bias,
                         int relu, int batches)
{
  long outs = (long)M * N;
  dim3 grid((unsigned)((outs + 127) / 128), 1, batches);
  small_gemm_kernel<<<grid, 128, 0, st>>>(A, lda, aB, B, ldb, bB, bt,
                                          Cf, Cbf, ldc, cB, M, N, K, alpha, bias, relu);
}

static void launch_convT(hipStream_t st, const float* W, unsigned short* Wt, int Kd, int Nd)
{
  long n = (long)Kd * Nd;
  convert_wt_kernel<<<(unsigned)((n + 255) / 256), 256, 0, st>>>(W, Wt, Kd, Nd);
}

extern "C" void kernel_launch(void* const* d_in, const int* in_sizes, int n_in,
                              void* d_out, int out_size, void* d_ws, size_t ws_size,
                              hipStream_t stream)
{
  (void)in_sizes; (void)n_in; (void)out_size; (void)ws_size;
  const int*   x          = (const int*)  d_in[0];
  const int*   target     = (const int*)  d_in[1];
  const float* in_emb     = (const float*)d_in[2];
  const float* out_emb    = (const float*)d_in[3];
  const float* enc_qkv_w  = (const float*)d_in[4];
  const float* enc_qkv_b  = (const float*)d_in[5];
  const float* enc_ln1_g  = (const float*)d_in[6];
  const float* enc_ln1_b  = (const float*)d_in[7];
  const float* enc_ffn1_w = (const float*)d_in[8];
  const float* enc_ffn1_b = (const float*)d_in[9];
  const float* enc_ffn2_w = (const float*)d_in[10];
  const float* enc_ffn2_b = (const float*)d_in[11];
  const float* enc_ln2_g  = (const float*)d_in[12];
  const float* enc_ln2_b  = (const float*)d_in[13];
  const float* dec_qkv1_w = (const float*)d_in[14];
  const float* dec_qkv1_b = (const float*)d_in[15];
  const float* dec_ln1_g  = (const float*)d_in[16];
  const float* dec_ln1_b  = (const float*)d_in[17];
  const float* dec_qkv2_w = (const float*)d_in[18];
  const float* dec_qkv2_b = (const float*)d_in[19];
  const float* dec_ln2_g  = (const float*)d_in[20];
  const float* dec_ln2_b  = (const float*)d_in[21];
  const float* dec_ffn1_w = (const float*)d_in[22];
  const float* dec_ffn1_b = (const float*)d_in[23];
  const float* dec_ffn2_w = (const float*)d_in[24];
  const float* dec_ffn2_b = (const float*)d_in[25];
  const float* dec_ln3_g  = (const float*)d_in[26];
  const float* dec_ln3_b  = (const float*)d_in[27];
  const float* out_w      = (const float*)d_in[28];
  const float* out_b      = (const float*)d_in[29];

  // ---- workspace layout (~64.1 MB) ---------------------------------------
  char* ws = (char*)d_ws;
  float*          e       = (float*)          (ws + ((size_t) 0u << 20)); // 8 MB  [B,S,D] fp32
  unsigned short* ebf     = (unsigned short*) (ws + ((size_t) 8u << 20)); // 4 MB  bf16
  float*          tmpf    = (float*)          (ws + ((size_t)12u << 20)); // 8 MB  attn/ffn out
  unsigned short* qbf     = (unsigned short*) (ws + ((size_t)20u << 20)); // 4 MB
  unsigned short* kbf     = (unsigned short*) (ws + ((size_t)24u << 20)); // 4 MB
  unsigned short* vtbf    = (unsigned short*) (ws + ((size_t)28u << 20)); // 4 MB V^T [D, B*S]
  unsigned short* h1bf    = (unsigned short*) (ws + ((size_t)32u << 20)); // 16 MB [B,S,DFF]
  float*          scoresf = (float*)          (ws + ((size_t)48u << 20)); // 8 MB  [H,S,S] per batch
  unsigned short* pbf     = (unsigned short*) (ws + ((size_t)56u << 20)); // 4 MB
  unsigned short* wtbuf   = (unsigned short*) (ws + ((size_t)60u << 20)); // 4 MB weight prepack
  char* db = ws + ((size_t)64u << 20);
  float*          dbuf  = (float*)          (db);           // 16 KB [B,1,D]
  unsigned short* dbf   = (unsigned short*) (db + 16384);   //  8 KB
  float*          dtmpf = (float*)          (db + 24576);   // 16 KB
  unsigned short* dq    = (unsigned short*) (db + 40960);   //  8 KB
  unsigned short* dk    = (unsigned short*) (db + 49152);   //  8 KB
  unsigned short* dv    = (unsigned short*) (db + 57344);   //  8 KB
  unsigned short* dh1   = (unsigned short*) (db + 65536);   // 32 KB [B,DFF]

  const int Mrows = BATCH * SEQ;    // 4096
  const long SD   = (long)SEQ * D_MODEL;
  const long SS   = (long)SEQ * SEQ;

  // ======================= ENCODER ========================================
  enc_embed_kernel<<<Mrows, 128, 0, stream>>>(x, in_emb, e, ebf);

  for (int l = 0; l < NLAYER; ++l) {
    // Q,K,V projections (bf16 outputs; V stored transposed -> fast PV path)
    for (int j = 0; j < 3; ++j) {
      launch_convT(stream, enc_qkv_w + (size_t)(l * 3 + j) * D_MODEL * D_MODEL,
                   wtbuf, D_MODEL, D_MODEL);
      if (j < 2) {
        launch_full(stream, ebf, D_MODEL, 0, wtbuf, D_MODEL, 0,
                    nullptr, (j == 0) ? qbf : kbf, D_MODEL, 0,
                    Mrows, D_MODEL, D_MODEL, 1.0f,
                    enc_qkv_b + (size_t)(l * 3 + j) * D_MODEL, 0, 1);
      } else {
        launch_full(stream, ebf, D_MODEL, 0, wtbuf, D_MODEL, 0,
                    nullptr, vtbf, Mrows, 0,        // V^T: [n][m], ldc = 4096
                    Mrows, D_MODEL, D_MODEL, 1.0f,
                    enc_qkv_b + (size_t)(l * 3 + j) * D_MODEL, 0, 1, /*cT=*/1);
      }
    }
    // attention, per batch (grid.z = heads)
    for (int b = 0; b < BATCH; ++b) {
      const unsigned short* qb = qbf + (size_t)b * SD;
      const unsigned short* kb = kbf + (size_t)b * SD;
      // scores[h,q,k] = q·k / sqrt(dk)
      launch_full(stream, qb, D_MODEL, DK, kb, D_MODEL, DK,
                  scoresf, nullptr, SEQ, SS, SEQ, SEQ, DK, 0.125f, nullptr, 0, NHEAD);
      softmax_mask_kernel<<<(NHEAD * SEQ + 3) / 4, 128, 0, stream>>>(
          scoresf, pbf, NHEAD * SEQ, SEQ, SEQ, 0);
      // out[q, h*64+n] = P @ V   (B = V^T rows n=h*64+d, contiguous in k=s)
      launch_full(stream, pbf, SEQ, SS,
                  vtbf + (size_t)b * SEQ, Mrows, (long)DK * Mrows,
                  tmpf + (size_t)b * SD, nullptr, D_MODEL, DK,
                  SEQ, DK, SEQ, 1.0f, nullptr, 0, NHEAD);
    }
    ln_residual_kernel<<<(Mrows + 3) / 4, 128, 0, stream>>>(
        tmpf, e, enc_ln1_g + (size_t)l * D_MODEL, enc_ln1_b + (size_t)l * D_MODEL,
        e, ebf, Mrows);
    // FFN
    launch_convT(stream, enc_ffn1_w + (size_t)l * D_MODEL * DFF_DIM, wtbuf, D_MODEL, DFF_DIM);
    launch_full(stream, ebf, D_MODEL, 0, wtbuf, D_MODEL, 0,
                nullptr, h1bf, DFF_DIM, 0, Mrows, DFF_DIM, D_MODEL, 1.0f,
                enc_ffn1_b + (size_t)l * DFF_DIM, 1, 1);
    launch_convT(stream, enc_ffn2_w + (size_t)l * DFF_DIM * D_MODEL, wtbuf, DFF_DIM, D_MODEL);
    launch_full(stream, h1bf, DFF_DIM, 0, wtbuf, DFF_DIM, 0,
                tmpf, nullptr, D_MODEL, 0, Mrows, D_MODEL, DFF_DIM, 1.0f,
                enc_ffn2_b + (size_t)l * D_MODEL, 0, 1);
    ln_residual_kernel<<<(Mrows + 3) / 4, 128, 0, stream>>>(
        e, tmpf, enc_ln2_g + (size_t)l * D_MODEL, enc_ln2_b + (size_t)l * D_MODEL,
        e, ebf, Mrows);
  }

  // ======================= DECODER (seq len 1) ============================
  dec_embed_kernel<<<BATCH, 128, 0, stream>>>(target, out_emb, dbuf, dbf);

  for (int l = 0; l < NLAYER; ++l) {
    // self-attention QKV (M = 8 rows -> scalar-dot kernel)
    for (int j = 0; j < 3; ++j) {
      launch_convT(stream, dec_qkv1_w + (size_t)(l * 3 + j) * D_MODEL * D_MODEL,
                   wtbuf, D_MODEL, D_MODEL);
      unsigned short* dst = (j == 0) ? dq : (j == 1) ? dk : dv;
      launch_small(stream, dbf, D_MODEL, 0, wtbuf, D_MODEL, 0, 1,
                   nullptr, dst, D_MODEL, 0, BATCH, D_MODEL, D_MODEL, 1.0f,
                   dec_qkv1_b + (size_t)(l * 3 + j) * D_MODEL, 0, 1);
    }
    // self scores: z = b*H+h, M=N=1, K=64
    launch_small(stream, dq, DK, DK, dk, DK, DK, 1,
                 scoresf, nullptr, 1, 1, 1, 1, DK, 0.125f, nullptr, 0, BATCH * NHEAD);
    softmax_mask_kernel<<<(BATCH * NHEAD + 3) / 4, 128, 0, stream>>>(
        scoresf, pbf, BATCH * NHEAD, 1, 1, 1);
    launch_small(stream, pbf, 1, 1, dv, DK, DK, 0,
                 dtmpf, nullptr, DK, DK, 1, DK, 1, 1.0f, nullptr, 0, BATCH * NHEAD);
    ln_residual_kernel<<<2, 128, 0, stream>>>(
        dtmpf, dbuf, dec_ln1_g + (size_t)l * D_MODEL, dec_ln1_b + (size_t)l * D_MODEL,
        dbuf, dbf, BATCH);

    // cross-attention: q from decoder (M=8), k/v from encoder output (M=4096)
    launch_convT(stream, dec_qkv2_w + (size_t)(l * 3 + 0) * D_MODEL * D_MODEL, wtbuf, D_MODEL, D_MODEL);
    launch_small(stream, dbf, D_MODEL, 0, wtbuf, D_MODEL, 0, 1,
                 nullptr, dq, D_MODEL, 0, BATCH, D_MODEL, D_MODEL, 1.0f,
                 dec_qkv2_b + (size_t)(l * 3 + 0) * D_MODEL, 0, 1);
    launch_convT(stream, dec_qkv2_w + (size_t)(l * 3 + 1) * D_MODEL * D_MODEL, wtbuf, D_MODEL, D_MODEL);
    launch_full(stream, ebf, D_MODEL, 0, wtbuf, D_MODEL, 0,
                nullptr, kbf, D_MODEL, 0, Mrows, D_MODEL, D_MODEL, 1.0f,
                dec_qkv2_b + (size_t)(l * 3 + 1) * D_MODEL, 0, 1);
    launch_convT(stream, dec_qkv2_w + (size_t)(l * 3 + 2) * D_MODEL * D_MODEL, wtbuf, D_MODEL, D_MODEL);
    launch_full(stream, ebf, D_MODEL, 0, wtbuf, D_MODEL, 0,
                nullptr, vtbf, Mrows, 0,
                Mrows, D_MODEL, D_MODEL, 1.0f,
                dec_qkv2_b + (size_t)(l * 3 + 2) * D_MODEL, 0, 1, /*cT=*/1);

    for (int b = 0; b < BATCH; ++b) {
      // scores[h, 0, k]: M=1, N=512, K=64
      launch_small(stream, dq + (size_t)b * D_MODEL, DK, DK,
                   kbf + (size_t)b * SD, D_MODEL, DK, 1,
                   scoresf, nullptr, SEQ, SEQ, 1, SEQ, DK, 0.125f, nullptr, 0, NHEAD);
      softmax_mask_kernel<<<2, 128, 0, stream>>>(scoresf, pbf, NHEAD, SEQ, 1, 0);
      launch_small(stream, pbf, SEQ, SEQ,
                   vtbf + (size_t)b * SEQ, Mrows, (long)DK * Mrows, 1,
                   dtmpf + (size_t)b * D_MODEL, nullptr, D_MODEL, DK,
                   1, DK, SEQ, 1.0f, nullptr, 0, NHEAD);
    }
    ln_residual_kernel<<<2, 128, 0, stream>>>(
        dtmpf, dbuf, dec_ln2_g + (size_t)l * D_MODEL, dec_ln2_b + (size_t)l * D_MODEL,
        dbuf, dbf, BATCH);

    // FFN (M = 8 -> scalar-dot kernel)
    launch_convT(stream, dec_ffn1_w + (size_t)l * D_MODEL * DFF_DIM, wtbuf, D_MODEL, DFF_DIM);
    launch_small(stream, dbf, D_MODEL, 0, wtbuf, D_MODEL, 0, 1,
                 nullptr, dh1, DFF_DIM, 0, BATCH, DFF_DIM, D_MODEL, 1.0f,
                 dec_ffn1_b + (size_t)l * DFF_DIM, 1, 1);
    launch_convT(stream, dec_ffn2_w + (size_t)l * DFF_DIM * D_MODEL, wtbuf, DFF_DIM, D_MODEL);
    launch_small(stream, dh1, DFF_DIM, 0, wtbuf, DFF_DIM, 0, 1,
                 dtmpf, nullptr, D_MODEL, 0, BATCH, D_MODEL, DFF_DIM, 1.0f,
                 dec_ffn2_b + (size_t)l * D_MODEL, 0, 1);
    ln_residual_kernel<<<2, 128, 0, stream>>>(
        dbuf, dtmpf, dec_ln3_g + (size_t)l * D_MODEL, dec_ln3_b + (size_t)l * D_MODEL,
        dbuf, dbf, BATCH);
  }

  // ======================= OUTPUT PROJECTION ==============================
  launch_convT(stream, out_w, wtbuf, D_MODEL, OUT_DIM);
  launch_small(stream, dbf, D_MODEL, 0, wtbuf, D_MODEL, 0, 1,
               (float*)d_out, nullptr, OUT_DIM, 0, BATCH, OUT_DIM, D_MODEL, 1.0f,
               out_b, 0, 1);
}
